// Graphormer_39384850104775
// MI455X (gfx1250) — matrix operand order repaired and checked by reading, not compiled, not run on hardware
//
#include <hip/hip_runtime.h>
#include <hip/hip_bf16.h>

// ---------------------------------------------------------------------------
// Types for WMMA
// ---------------------------------------------------------------------------
typedef __attribute__((ext_vector_type(16))) __bf16          v16bf;
typedef __attribute__((ext_vector_type(8)))  float           v8f;
typedef __attribute__((ext_vector_type(16))) unsigned short  us16;
typedef __attribute__((ext_vector_type(8)))  unsigned short  us8;

union BFrag { us16 u; v16bf b; };

__device__ inline v8f wmma_bf16(v16bf a, v16bf b, v8f c) {
    return __builtin_amdgcn_wmma_f32_16x16x32_bf16(false, a, false, b,
                                                   (short)0, c, false, false);
}

// f32 -> bf16 round-to-nearest-even
__device__ inline unsigned short f2bf(float f) {
    unsigned int u = __float_as_uint(f);
    u += 0x7FFFu + ((u >> 16) & 1u);
    return (unsigned short)(u >> 16);
}

__device__ inline float gelu_tanh(float x) {
    const float c0 = 0.7978845608028654f, c1 = 0.044715f;
    float t = tanhf(c0 * (x + c1 * x * x * x));
    return 0.5f * x * (1.0f + t);
}

// A-fragment (16x32 bf16): lane lr holds row M=lr; element e -> K = e + 16*lh
__device__ inline v16bf loadA16(const unsigned short* rowp, int lh) {
    us8 a0 = *(const us8*)(rowp + 16 * lh);
    us8 a1 = *(const us8*)(rowp + 16 * lh + 8);
    BFrag f;
#pragma unroll
    for (int e = 0; e < 8; ++e) { f.u[e] = a0[e]; f.u[e + 8] = a1[e]; }
    return f.b;
}

// B-fragment (32x16 bf16): lane lr holds col N=lr; 16 contiguous K starting at 16*lh
__device__ inline v16bf loadB16(const unsigned short* p) {
    us8 b0 = *(const us8*)p;
    us8 b1 = *(const us8*)(p + 8);
    BFrag f;
#pragma unroll
    for (int e = 0; e < 8; ++e) { f.u[e] = b0[e]; f.u[e + 8] = b1[e]; }
    return f.b;
}

// gfx1250 async copy: 16B per lane, global -> LDS, tracked by ASYNCcnt
__device__ inline void async_b128(void* lds, const void* g) {
    unsigned int off = (unsigned int)(unsigned long long)lds;   // LDS offset = low 32 bits
    unsigned long long ga = (unsigned long long)g;
    asm volatile("global_load_async_to_lds_b128 %0, %1, off"
                 :: "v"(off), "v"(ga) : "memory");
}
__device__ inline void wait_async0() {
    asm volatile("s_wait_asynccnt 0" ::: "memory");
}

// ---------------------------------------------------------------------------
// Problem constants
// ---------------------------------------------------------------------------
constexpr int Nn = 2048, Ee = 32768;
constexpr int DIN = 128, Dd = 512, EIN = 64, EDd = 128;
constexpr int Hh = 8, FFf = 2048, LPp = 5, LYRl = 2, OUTo = 512;

// ---------------------------------------------------------------------------
// Generic WMMA GEMM: C = epi(A[MxK] @ B[KxNc] + bias)
// Block tile 128x64, BK=32, 256 threads (8 waves). M%128==0, Nc%64==0, K%32==0.
// Batched via blockIdx.z (strides in elements, bsC in BYTES).
// ---------------------------------------------------------------------------
template <bool OBF16, bool TRANS, bool GELU, bool RESID>
__global__ __launch_bounds__(256) void gemm_k(
    const float* __restrict__ A, const float* __restrict__ B,
    const float* __restrict__ bias, const float* __restrict__ resid,
    void* __restrict__ C, int M, int Nc, int K,
    long bsA, long bsB, long bsBias, long bsC)
{
    __shared__ unsigned short As[128 * 32];   // row-major, stride 32
    __shared__ unsigned short Bs[64 * 32];    // col-major: Bs[col*32 + k]

    const int tid  = threadIdx.x;
    const int wave = tid >> 5, lane = tid & 31;
    const int lh   = lane >> 4, lr = lane & 15;
    const int n0   = blockIdx.x * 64;
    const int row0 = blockIdx.y * 128;
    const int z    = blockIdx.z;

    A    += (long)z * bsA;
    B    += (long)z * bsB;
    bias += (long)z * bsBias;
    char* Cb = (char*)C + (long)z * bsC;

    v8f acc[4];
#pragma unroll
    for (int c = 0; c < 4; ++c) acc[c] = (v8f){0.f,0.f,0.f,0.f,0.f,0.f,0.f,0.f};

    for (int k0 = 0; k0 < K; k0 += 32) {
        __syncthreads();
        // stage A tile 128x32 (f32 -> bf16)
        for (int i = tid; i < 1024; i += 256) {
            int r = i >> 3, c4 = (i & 7) << 2;
            const float4 f = *(const float4*)(A + (size_t)(row0 + r) * K + k0 + c4);
            As[r * 32 + c4 + 0] = f2bf(f.x);
            As[r * 32 + c4 + 1] = f2bf(f.y);
            As[r * 32 + c4 + 2] = f2bf(f.z);
            As[r * 32 + c4 + 3] = f2bf(f.w);
        }
        // stage B tile 32x64, transposed into column-major
        for (int i = tid; i < 512; i += 256) {
            int r = i >> 4, c4 = (i & 15) << 2;
            const float4 f = *(const float4*)(B + (size_t)(k0 + r) * Nc + n0 + c4);
            Bs[(c4 + 0) * 32 + r] = f2bf(f.x);
            Bs[(c4 + 1) * 32 + r] = f2bf(f.y);
            Bs[(c4 + 2) * 32 + r] = f2bf(f.z);
            Bs[(c4 + 3) * 32 + r] = f2bf(f.w);
        }
        __syncthreads();

        v16bf a = loadA16(&As[(wave * 16 + lr) * 32], lh);
#pragma unroll
        for (int c = 0; c < 4; ++c) {
            v16bf b = loadB16(&Bs[(c * 16 + lr) * 32 + 16 * lh]);
            acc[c] = wmma_bf16(a, b, acc[c]);
        }
    }

    // epilogue: C element (k,lane): row = k + 8*lh, col = lr
#pragma unroll
    for (int c = 0; c < 4; ++c) {
#pragma unroll
        for (int k = 0; k < 8; ++k) {
            int r  = row0 + wave * 16 + k + 8 * lh;
            int cc = n0 + c * 16 + lr;
            float v = acc[c][k] + bias[cc];
            if (GELU)  v = gelu_tanh(v);
            if (RESID) v += resid[(size_t)r * Nc + cc];
            if (OBF16) {
                unsigned short* Co = (unsigned short*)Cb;
                if (TRANS) Co[(size_t)cc * M + r] = f2bf(v);
                else       Co[(size_t)r * Nc + cc] = f2bf(v);
            } else {
                ((float*)Cb)[(size_t)r * Nc + cc] = v;
            }
        }
    }
}

// ---------------------------------------------------------------------------
// LayerNorm over D=512, one row per block (256 threads)
// ---------------------------------------------------------------------------
__global__ __launch_bounds__(256) void ln_k(const float* __restrict__ xin,
                                            const float* __restrict__ g,
                                            const float* __restrict__ b,
                                            float* __restrict__ out)
{
    const int row = blockIdx.x, tid = threadIdx.x;
    const int wave = tid >> 5, lane = tid & 31;
    __shared__ float s_sum[8], s_sq[8], s_mv[2];

    float v0 = xin[(size_t)row * 512 + tid];
    float v1 = xin[(size_t)row * 512 + 256 + tid];
    float s = v0 + v1, q = v0 * v0 + v1 * v1;
#pragma unroll
    for (int o = 16; o > 0; o >>= 1) {
        s += __shfl_xor(s, o, 32);
        q += __shfl_xor(q, o, 32);
    }
    if (lane == 0) { s_sum[wave] = s; s_sq[wave] = q; }
    __syncthreads();
    if (tid == 0) {
        float S = 0.f, Q = 0.f;
        for (int i = 0; i < 8; ++i) { S += s_sum[i]; Q += s_sq[i]; }
        float m = S * (1.0f / 512.0f);
        float var = Q * (1.0f / 512.0f) - m * m;
        s_mv[0] = m;
        s_mv[1] = rsqrtf(var + 1e-5f);
    }
    __syncthreads();
    float m = s_mv[0], r = s_mv[1];
    out[(size_t)row * 512 + tid]       = (v0 - m) * r * g[tid]       + b[tid];
    out[(size_t)row * 512 + 256 + tid] = (v1 - m) * r * g[256 + tid] + b[256 + tid];
}

// x += z_in[clip(in_deg)] + z_out[clip(out_deg)]
__global__ void deg_add_k(float* __restrict__ x, const float* __restrict__ zi,
                          const float* __restrict__ zo,
                          const int* __restrict__ idg, const int* __restrict__ odg)
{
    int idx = blockIdx.x * 256 + threadIdx.x;        // N*D
    int n = idx >> 9, d = idx & 511;
    int a = idg[n]; a = a < 0 ? 0 : (a > 63 ? 63 : a);
    int c = odg[n]; c = c < 0 ? 0 : (c > 63 ? 63 : c);
    x[idx] += zi[a * 512 + d] + zo[c * 512 + d];
}

// bmat[i,j] = lens>0 ? b_spatial[min(lens,5)-1] : 0
__global__ void bmat_k(const int* __restrict__ npl, const float* __restrict__ bsp,
                       float* __restrict__ bm)
{
    int idx = blockIdx.x * 256 + threadIdx.x;        // N*N
    int len = npl[idx];
    float v = 0.f;
    if (len > 0) { int c = len < 5 ? len : 5; v = bsp[c - 1]; }
    bm[idx] = v;
}

// wbuf[e, h*5+p] = dot(ea[e,:], edge_vec[l,h,p,:])  (K=128)
__global__ void edgew_k(const float* __restrict__ ea, const float* __restrict__ ev,
                        float* __restrict__ wb)
{
    int idx = blockIdx.x * 256 + threadIdx.x;        // E*40
    int e = idx / 40, hp = idx % 40;
    const float* ar = ea + (size_t)e * 128;
    const float* br = ev + (size_t)hp * 128;
    float s = 0.f;
#pragma unroll 4
    for (int k = 0; k < 128; ++k) s += ar[k] * br[k];
    wb[idx] = s;
}

// ---------------------------------------------------------------------------
// Flash attention with on-the-fly graph bias.
// grid (H, N/16); block 256 (8 waves). Each block: 16 query rows of one head.
// q,k: (H,N,512) bf16 row-major.  vT: (H,512,N) bf16.  o: (N, H*512) f32.
// Wave w: S-phase -> C tile jt=w>>2 over d-slice (w&3)*128;
//         O-phase -> owns output d-columns [64w, 64w+64).
// ---------------------------------------------------------------------------
__global__ __launch_bounds__(256) void attn_k(
    const unsigned short* __restrict__ qg, const unsigned short* __restrict__ kg,
    const unsigned short* __restrict__ vtg,
    const float* __restrict__ bmat, const float* __restrict__ wbuf,
    const int* __restrict__ ept, const int* __restrict__ epl,
    float* __restrict__ og, float scale)
{
    const int h  = blockIdx.x;
    const int i0 = blockIdx.y * 16;
    const int tid = threadIdx.x;
    const int wave = tid >> 5, lane = tid & 31;
    const int lh = lane >> 4, lr = lane & 15;

    __shared__ unsigned short Qs[16 * 512];   // 16KB
    __shared__ unsigned short Ks[32 * 512];   // 32KB (row j, col d)
    __shared__ unsigned short Vs[512 * 32];   // 32KB (row d, col j)
    __shared__ float Sp[8 * 256];             // per-wave S partials
    __shared__ float Sf[512];                 // 16x32 scores / probs
    __shared__ float m_s[16], l_s[16], corr_s[16];

    const unsigned short* qh = qg  + (size_t)h * Nn * Dd;
    const unsigned short* kh = kg  + (size_t)h * Nn * Dd;
    const unsigned short* vh = vtg + (size_t)h * Dd * Nn;

    // Q tile: 16 rows x 512 (async, 16B/lane)
    for (int i = tid; i < 1024; i += 256) {
        int r = i >> 6, c8 = (i & 63) << 3;
        async_b128(&Qs[r * 512 + c8], qh + (size_t)(i0 + r) * 512 + c8);
    }
    wait_async0();
    if (tid < 16) { m_s[tid] = -3.0e38f; l_s[tid] = 0.f; }
    __syncthreads();

    v8f acc[4];
#pragma unroll
    for (int c = 0; c < 4; ++c) acc[c] = (v8f){0.f,0.f,0.f,0.f,0.f,0.f,0.f,0.f};

    const int jt = wave >> 2;            // which 16-col S tile
    const int dbase = (wave & 3) * 128;  // d-slice for S partial

    for (int j0 = 0; j0 < Nn; j0 += 32) {
        __syncthreads();   // prior readers of Ks/Vs/Sf are done
        for (int i = tid; i < 2048; i += 256) {      // K tile 32x512
            int r = i >> 6, c8 = (i & 63) << 3;
            async_b128(&Ks[r * 512 + c8], kh + (size_t)(j0 + r) * 512 + c8);
        }
        for (int i = tid; i < 2048; i += 256) {      // V tile 512x32 (from vT)
            int d = i >> 2, j8 = (i & 3) << 3;
            async_b128(&Vs[d * 32 + j8], vh + (size_t)d * Nn + j0 + j8);
        }
        wait_async0();
        __syncthreads();

        // --- S partials: Q(16 x 128-slice) @ K^T ---
        v8f sc = (v8f){0.f,0.f,0.f,0.f,0.f,0.f,0.f,0.f};
#pragma unroll
        for (int dc = 0; dc < 4; ++dc) {
            int koff = dbase + dc * 32;
            v16bf a = loadA16(&Qs[lr * 512 + koff], lh);
            v16bf b = loadB16(&Ks[(jt * 16 + lr) * 512 + koff + 16 * lh]);
            sc = wmma_bf16(a, b, sc);
        }
#pragma unroll
        for (int k = 0; k < 8; ++k)
            Sp[wave * 256 + (k + 8 * lh) * 16 + lr] = sc[k];
        __syncthreads();

        // --- reduce partials + spatial / edge-path bias ---
        for (int e = tid; e < 512; e += 256) {
            int t2 = e >> 8, rr = (e >> 4) & 15, cc = e & 15;
            float v = Sp[(t2 * 4 + 0) * 256 + rr * 16 + cc]
                    + Sp[(t2 * 4 + 1) * 256 + rr * 16 + cc]
                    + Sp[(t2 * 4 + 2) * 256 + rr * 16 + cc]
                    + Sp[(t2 * 4 + 3) * 256 + rr * 16 + cc];
            v *= scale;
            int gi = i0 + rr, gj = j0 + t2 * 16 + cc;
            long pij = (long)gi * Nn + gj;
            v += bmat[pij];
            int pl = epl[pij];
            float cs = 0.f;
#pragma unroll
            for (int p = 0; p < 5; ++p)
                if (pl > p)
                    cs += wbuf[(size_t)ept[pij * 5 + p] * 40 + h * 5 + p];
            if (pl > 0) v += cs / (float)pl;
            Sf[rr * 32 + t2 * 16 + cc] = v;
        }
        __syncthreads();

        // --- online softmax stats (one thread per row) ---
        if (tid < 16) {
            float m_old = m_s[tid], mx = m_old;
#pragma unroll 8
            for (int j = 0; j < 32; ++j) mx = fmaxf(mx, Sf[tid * 32 + j]);
            float corr = __expf(m_old - mx);
            float l = l_s[tid] * corr;
#pragma unroll 8
            for (int j = 0; j < 32; ++j) {
                float ev = __expf(Sf[tid * 32 + j] - mx);
                Sf[tid * 32 + j] = ev;
                l += ev;
            }
            m_s[tid] = mx; l_s[tid] = l; corr_s[tid] = corr;
        }
        __syncthreads();

        // --- rescale O, then O += P @ V ---
        float cr[8];
#pragma unroll
        for (int k = 0; k < 8; ++k) cr[k] = corr_s[k + 8 * lh];
#pragma unroll
        for (int c = 0; c < 4; ++c)
#pragma unroll
            for (int k = 0; k < 8; ++k) acc[c][k] *= cr[k];

        BFrag pf;                                // P as 16x32 bf16 A-fragment
#pragma unroll
        for (int e = 0; e < 16; ++e) {
            int jc = e + 16 * lh;
            pf.u[e] = f2bf(Sf[lr * 32 + jc]);
        }
#pragma unroll
        for (int c = 0; c < 4; ++c) {
            v16bf b = loadB16(&Vs[(wave * 64 + c * 16 + lr) * 32 + 16 * lh]);
            acc[c] = wmma_bf16(pf.b, b, acc[c]);
        }
    }

    // write O / l  into (N, H*512) layout
#pragma unroll
    for (int c = 0; c < 4; ++c)
#pragma unroll
        for (int k = 0; k < 8; ++k) {
            int r = k + 8 * lh;
            float o = acc[c][k] / l_s[r];
            og[(size_t)(i0 + r) * (Hh * Dd) + h * Dd + wave * 64 + c * 16 + lr] = o;
        }
}

// ---------------------------------------------------------------------------
// Host orchestration
// ---------------------------------------------------------------------------
extern "C" void kernel_launch(void* const* d_in, const int* in_sizes, int n_in,
                              void* d_out, int out_size, void* d_ws, size_t ws_size,
                              hipStream_t stream)
{
    (void)in_sizes; (void)n_in; (void)out_size; (void)ws_size;

    const float* x_in   = (const float*)d_in[0];
    const float* eattr  = (const float*)d_in[1];
    const int*   in_deg = (const int*)d_in[3];
    const int*   out_deg= (const int*)d_in[4];
    const int*   npl    = (const int*)d_in[5];
    const int*   ept    = (const int*)d_in[6];
    const int*   epl    = (const int*)d_in[7];
    const float* niw    = (const float*)d_in[8];
    const float* nib    = (const float*)d_in[9];
    const float* eiw    = (const float*)d_in[10];
    const float* eib    = (const float*)d_in[11];
    const float* z_in   = (const float*)d_in[12];
    const float* z_out  = (const float*)d_in[13];
    const float* bsp    = (const float*)d_in[14];
    const float* ln1g   = (const float*)d_in[15];
    const float* ln1b   = (const float*)d_in[16];
    const float* ln2g   = (const float*)d_in[17];
    const float* ln2b   = (const float*)d_in[18];
    const float* wq     = (const float*)d_in[19];
    const float* bq     = (const float*)d_in[20];
    const float* wk     = (const float*)d_in[21];
    const float* bk     = (const float*)d_in[22];
    const float* wv     = (const float*)d_in[23];
    const float* bv     = (const float*)d_in[24];
    const float* wo     = (const float*)d_in[25];
    const float* bo     = (const float*)d_in[26];
    const float* evec   = (const float*)d_in[27];
    const float* f1w    = (const float*)d_in[28];
    const float* f1b    = (const float*)d_in[29];
    const float* f2w    = (const float*)d_in[30];
    const float* f2b    = (const float*)d_in[31];
    const float* now    = (const float*)d_in[32];
    const float* nob    = (const float*)d_in[33];

    // workspace carve (256B aligned regions)
    char* p = (char*)d_ws;
    auto take = [&](size_t bytes) {
        char* r = p;
        p += (bytes + 255) & ~(size_t)255;
        return r;
    };
    float* xb  = (float*)take((size_t)Nn * Dd * 4);
    float* hb  = (float*)take((size_t)Nn * Dd * 4);
    float* eab = (float*)take((size_t)Ee * EDd * 4);
    float* bmb = (float*)take((size_t)Nn * Nn * 4);
    float* wbf = (float*)take((size_t)Ee * Hh * LPp * 4);
    float* ob  = (float*)take((size_t)Nn * Hh * Dd * 4);
    float* fb  = (float*)take((size_t)Nn * FFf * 4);
    unsigned short* qb = (unsigned short*)take((size_t)Hh * Nn * Dd * 2);
    unsigned short* kb = (unsigned short*)take((size_t)Hh * Nn * Dd * 2);
    unsigned short* vb = (unsigned short*)take((size_t)Hh * Dd * Nn * 2);

    const float scale = 1.0f / sqrtf((float)Dd);
    const long qkvC = (long)Nn * Dd * 2;   // per-head C stride (bytes)

    // 1) x = x_in @ node_in_w + b
    gemm_k<false,false,false,false><<<dim3(Dd/64, Nn/128, 1), 256, 0, stream>>>(
        x_in, niw, nib, nullptr, xb, Nn, Dd, DIN, 0, 0, 0, 0);
    // 2) += degree embeddings
    deg_add_k<<<(Nn * Dd) / 256, 256, 0, stream>>>(xb, z_in, z_out, in_deg, out_deg);
    // 3) ea = edge_attr @ edge_in_w + b
    gemm_k<false,false,false,false><<<dim3(EDd/64, Ee/128, 1), 256, 0, stream>>>(
        eattr, eiw, eib, nullptr, eab, Ee, EDd, EIN, 0, 0, 0, 0);
    // 4) spatial bias matrix
    bmat_k<<<(Nn * (size_t)Nn) / 256, 256, 0, stream>>>(npl, bsp, bmb);

    for (int l = 0; l < LYRl; ++l) {
        // h = LN1(x)
        ln_k<<<Nn, 256, 0, stream>>>(xb, ln1g + l * Dd, ln1b + l * Dd, hb);
        // edge-path weights  w[e,h,p]
        edgew_k<<<(Ee * Hh * LPp) / 256, 256, 0, stream>>>(
            eab, evec + (size_t)l * Hh * LPp * EDd, wbf);
        // batched per-head Q,K (bf16 row-major) and V (bf16 transposed)
        gemm_k<true,false,false,false><<<dim3(Dd/64, Nn/128, Hh), 256, 0, stream>>>(
            hb, wq + (size_t)l * Hh * Dd * Dd, bq + (size_t)l * Hh * Dd, nullptr, qb,
            Nn, Dd, Dd, 0, (long)Dd * Dd, Dd, qkvC);
        gemm_k<true,false,false,false><<<dim3(Dd/64, Nn/128, Hh), 256, 0, stream>>>(
            hb, wk + (size_t)l * Hh * Dd * Dd, bk + (size_t)l * Hh * Dd, nullptr, kb,
            Nn, Dd, Dd, 0, (long)Dd * Dd, Dd, qkvC);
        gemm_k<true,true,false,false><<<dim3(Dd/64, Nn/128, Hh), 256, 0, stream>>>(
            hb, wv + (size_t)l * Hh * Dd * Dd, bv + (size_t)l * Hh * Dd, nullptr, vb,
            Nn, Dd, Dd, 0, (long)Dd * Dd, Dd, qkvC);
        // attention with graph bias -> o (N, H*D)
        attn_k<<<dim3(Hh, Nn / 16), 256, 0, stream>>>(
            qb, kb, vb, bmb, wbf, ept, epl, ob, scale);
        // x = x + o @ wo + bo
        gemm_k<false,false,false,true><<<dim3(Dd/64, Nn/128, 1), 256, 0, stream>>>(
            ob, wo + (size_t)l * (Hh * Dd) * Dd, bo + (size_t)l * Dd, xb, xb,
            Nn, Dd, Hh * Dd, 0, 0, 0, 0);
        // h = LN2(x)
        ln_k<<<Nn, 256, 0, stream>>>(xb, ln2g + l * Dd, ln2b + l * Dd, hb);
        // f = gelu(h @ ff1 + b1)
        gemm_k<false,false,true,false><<<dim3(FFf/64, Nn/128, 1), 256, 0, stream>>>(
            hb, f1w + (size_t)l * Dd * FFf, f1b + (size_t)l * FFf, nullptr, fb,
            Nn, FFf, Dd, 0, 0, 0, 0);
        // x = x + f @ ff2 + b2
        gemm_k<false,false,false,true><<<dim3(Dd/64, Nn/128, 1), 256, 0, stream>>>(
            fb, f2w + (size_t)l * FFf * Dd, f2b + (size_t)l * Dd, xb, xb,
            Nn, Dd, FFf, 0, 0, 0, 0);
    }

    // out = x @ node_out_w + b
    gemm_k<false,false,false,false><<<dim3(OUTo/64, Nn/128, 1), 256, 0, stream>>>(
        xb, now, nob, nullptr, (float*)d_out, Nn, OUTo, Dd, 0, 0, 0, 0);
}
